// DecoderBase_16509854286560
// MI455X (gfx1250) — matrix operand order repaired and checked by reading, not compiled, not run on hardware
//
#include <hip/hip_runtime.h>
#include <hip/hip_bf16.h>

// Problem constants (from reference)
#define NB   8
#define TI   200
#define TO   100
#define JD   512
#define VOC  1024
#define ROWS (NB * TI * TO)   // 160000, divisible by 64

typedef __attribute__((ext_vector_type(16))) __bf16 v16bf;
typedef __attribute__((ext_vector_type(8)))  float  v8f;

union ABFrag {
    uint4 q[2];   // two 16-byte halves
    v16bf v;      // 16 bf16 = 32 bytes
};

__device__ __forceinline__ unsigned short f2bf(float f) {
    // round-to-nearest-even f32 -> bf16 (inputs are finite here)
    unsigned u = __float_as_uint(f);
    unsigned r = u + 0x7FFFu + ((u >> 16) & 1u);
    return (unsigned short)(r >> 16);
}

__device__ __forceinline__ float fast_tanh(float x) {
#if __has_builtin(__builtin_amdgcn_tanhf)
    return __builtin_amdgcn_tanhf(x);
#elif __has_builtin(__builtin_amdgcn_tanh_f32)
    return __builtin_amdgcn_tanh_f32(x);
#else
    return tanhf(x);
#endif
}

// ---------------------------------------------------------------------------
// Projection: Y[r][j] = dot(X[r][:], W[j][:]) + bias[j], K = N = 512, f32.
// One block = 8 rows staged in LDS; each thread computes 2 columns x 8 rows.
// ---------------------------------------------------------------------------
__global__ __launch_bounds__(256) void proj_kernel(
    const float* __restrict__ X, const float* __restrict__ W,
    const float* __restrict__ bias, float* __restrict__ Y)
{
    __shared__ __align__(16) float xs[8 * JD];   // 16 KB

    const int tid = threadIdx.x;
    const int r0  = blockIdx.x * 8;

    const float4* X4 = (const float4*)(X + (size_t)r0 * JD);
    float4* xs4 = (float4*)xs;
#pragma unroll
    for (int i = 0; i < 4; ++i)
        xs4[tid + i * 256] = X4[tid + i * 256];
    __syncthreads();

    const float4* W4 = (const float4*)W;
#pragma unroll
    for (int jj = 0; jj < 2; ++jj) {
        const int j = tid + jj * 256;
        const float bv = bias ? bias[j] : 0.0f;
        float acc[8];
#pragma unroll
        for (int r = 0; r < 8; ++r) acc[r] = 0.0f;

        for (int k4 = 0; k4 < JD / 4; ++k4) {
            const float4 w = W4[(size_t)j * (JD / 4) + k4];
#pragma unroll
            for (int r = 0; r < 8; ++r) {
                const float4 x = xs4[r * (JD / 4) + k4];  // LDS broadcast
                acc[r] += x.x * w.x + x.y * w.y + x.z * w.z + x.w * w.w;
            }
        }
#pragma unroll
        for (int r = 0; r < 8; ++r)
            Y[(size_t)(r0 + r) * JD + j] = acc[r] + bv;
    }
}

// ---------------------------------------------------------------------------
// Convert W_out f32 -> bf16 (row-major [VOC][JD])
// ---------------------------------------------------------------------------
__global__ __launch_bounds__(256) void cvt_bf16_kernel(
    const float* __restrict__ src, unsigned short* __restrict__ dst, int n)
{
    int i = (blockIdx.x * 256 + threadIdx.x) * 4;
    if (i + 3 < n) {
        float4 v = *(const float4*)(src + i);
        dst[i + 0] = f2bf(v.x);
        dst[i + 1] = f2bf(v.y);
        dst[i + 2] = f2bf(v.z);
        dst[i + 3] = f2bf(v.w);
    }
}

// ---------------------------------------------------------------------------
// Fused joint kernel:
//   block = 64 output rows x all 1024 vocab cols.
//   Stage act = tanh(e_row + d_row) for 64 rows once into LDS (bf16).
//   Each wave: 4 row-strips x 4 col-tiles per chunk (16 accumulators),
//   2 chunks -> 128 cols per wave, 8 waves -> 1024 cols.
//   Every B fragment loaded is reused by 4 wmmas (4 row strips): W_out L2
//   traffic = 2500 blocks x 1 MB = 2.5 GB; output streamed with NT stores.
// ---------------------------------------------------------------------------
__global__ __launch_bounds__(256) void joint_kernel(
    const float* __restrict__ eproj,          // [NB*TI][JD] f32
    const float* __restrict__ dproj,          // [NB*TO][JD] f32 (bias folded)
    const unsigned short* __restrict__ wbf,   // [VOC][JD] bf16
    float* __restrict__ out)                  // [ROWS][VOC] f32
{
    constexpr int PAD = 8;                    // 16B pad -> rotate LDS banks
    __shared__ __align__(16) unsigned short act[64][JD + PAD];  // ~65 KB

    const int tid   = threadIdx.x;
    const int block = blockIdx.x;             // 0..ROWS/64-1

    // ---- stage activations: thread (r = tid/4) handles 128 K-elements ----
    {
        const int r  = tid >> 2;              // 0..63
        const int kc = (tid & 3) << 7;        // 0,128,256,384
        const int g   = block * 64 + r;       // global (n,t,u) row
        const int n   = g / (TI * TO);
        const int rem = g - n * (TI * TO);
        const int t   = rem / TO;
        const int u   = rem - t * TO;
        const float4* e4 = (const float4*)(eproj + (size_t)(n * TI + t) * JD + kc);
        const float4* d4 = (const float4*)(dproj + (size_t)(n * TO + u) * JD + kc);
#pragma unroll
        for (int i = 0; i < 32; ++i) {
            const float4 ev = e4[i];
            const float4 dv = d4[i];
            act[r][kc + i * 4 + 0] = f2bf(fast_tanh(ev.x + dv.x));
            act[r][kc + i * 4 + 1] = f2bf(fast_tanh(ev.y + dv.y));
            act[r][kc + i * 4 + 2] = f2bf(fast_tanh(ev.z + dv.z));
            act[r][kc + i * 4 + 3] = f2bf(fast_tanh(ev.w + dv.w));
        }
    }
    __syncthreads();

    // ---- WMMA phase ----
    const int lane = tid & 31;
    const int wv   = tid >> 5;                // 0..7
    const int nlo  = lane & 15;
    const int hi   = (lane >> 4) & 1;

    // A fragment (16x32 bf16): lane<16 -> K 0-7 & 16-23; lane>=16 -> K 8-15 & 24-31
    const int a_koff = hi * 8;
    // B fragment (32x16 bf16): lane<16 -> K 0-7 & 16-23; lane>=16 -> K 8-15 & 24-31; n = lane%16
    const int b_koff = hi * 8;

    const size_t base_row = (size_t)block * 64;

#pragma unroll
    for (int chunk = 0; chunk < 2; ++chunk) {
        const int colbase = wv * 128 + chunk * 64;   // 4 col tiles of 16

        v8f acc[4][4];
#pragma unroll
        for (int s = 0; s < 4; ++s)
#pragma unroll
            for (int ct = 0; ct < 4; ++ct)
                acc[s][ct] = (v8f){0,0,0,0,0,0,0,0};

        for (int kk = 0; kk < JD / 32; ++kk) {       // 16 K-steps
            const int k0 = kk * 32;

            ABFrag a[4];
#pragma unroll
            for (int s = 0; s < 4; ++s) {
                a[s].q[0] = *(const uint4*)&act[s * 16 + nlo][k0 + a_koff];
                a[s].q[1] = *(const uint4*)&act[s * 16 + nlo][k0 + a_koff + 16];
            }

            ABFrag b[4];
#pragma unroll
            for (int ct = 0; ct < 4; ++ct) {
                const unsigned short* bp =
                    wbf + (size_t)(colbase + ct * 16 + nlo) * JD + k0 + b_koff;
                b[ct].q[0] = *(const uint4*)bp;
                b[ct].q[1] = *(const uint4*)(bp + 16);
            }

#pragma unroll
            for (int ct = 0; ct < 4; ++ct)           // ct outer: partial waits
#pragma unroll
                for (int s = 0; s < 4; ++s)
                    acc[s][ct] = __builtin_amdgcn_wmma_f32_16x16x32_bf16(
                        false, a[s].v, false, b[ct].v, (short)0, acc[s][ct],
                        false, false);
        }

        // ---- store C (non-temporal: output is streamed, never re-read) ----
        // VGPR v -> row (strip*16 + v + hi*8), col = tile*16 + nlo
#pragma unroll
        for (int s = 0; s < 4; ++s) {
#pragma unroll
            for (int ct = 0; ct < 4; ++ct) {
                const int col = colbase + ct * 16 + nlo;
#pragma unroll
                for (int v = 0; v < 8; ++v) {
                    const size_t row = base_row + s * 16 + v + hi * 8;
                    __builtin_nontemporal_store(acc[s][ct][v],
                                                out + row * VOC + col);
                }
            }
        }
    }
}

extern "C" void kernel_launch(void* const* d_in, const int* in_sizes, int n_in,
                              void* d_out, int out_size, void* d_ws, size_t ws_size,
                              hipStream_t stream) {
    const float* enc_out = (const float*)d_in[0];   // (8,200,512)
    const float* dec_out = (const float*)d_in[1];   // (8,100,512)
    const float* W_enc   = (const float*)d_in[2];   // (512,512)
    const float* W_dec   = (const float*)d_in[3];   // (512,512)
    const float* b_dec   = (const float*)d_in[4];   // (512,)
    const float* W_out   = (const float*)d_in[5];   // (1024,512)
    float* out = (float*)d_out;                     // (8,200,100,1024)

    char* ws = (char*)d_ws;
    float*          eproj = (float*)ws;                                   // 1600*512*4
    float*          dproj = (float*)(ws + (size_t)1600 * JD * 4);         //  800*512*4
    unsigned short* wbf   = (unsigned short*)(ws + (size_t)2400 * JD * 4);// 1024*512*2

    // e = enc @ W_enc^T (no bias): 1600 rows / 8 per block
    proj_kernel<<<200, 256, 0, stream>>>(enc_out, W_enc, nullptr, eproj);
    // d = dec @ W_dec^T + b_dec: 800 rows / 8 per block
    proj_kernel<<<100, 256, 0, stream>>>(dec_out, W_dec, b_dec, dproj);
    // W_out -> bf16
    cvt_bf16_kernel<<<(VOC * JD) / (256 * 4), 256, 0, stream>>>(W_out, wbf, VOC * JD);
    // fused tanh-joint + vocab GEMM
    joint_kernel<<<ROWS / 64, 256, 0, stream>>>(eproj, dproj, wbf, out);
}
